// RNN_29463475650831
// MI455X (gfx1250) — matrix operand order repaired and checked by reading, not compiled, not run on hardware
//
#include <hip/hip_runtime.h>

// CDNA5 / gfx1250: wave32, WMMA 16x16x32 f16->f32.
typedef __attribute__((ext_vector_type(16))) _Float16     v16h;
typedef __attribute__((ext_vector_type(8)))  float        v8f;
typedef __attribute__((ext_vector_type(8)))  unsigned int v8u;

union UA { _Float16 h[16]; unsigned int u[8]; v16h v; v8u uv; };
union UC { float f[8];     v8f  v; };

static __device__ __forceinline__ v8f wmma_f16(v16h a, v16h b, v8f c) {
  return __builtin_amdgcn_wmma_f32_16x16x32_f16(false, a, false, b, (short)0, c, false, false);
}

// One v_cvt_pk_rtz_f16_f32: pack two f32 into one dword of f16 pairs.
// Note: the builtin returns an __fp16 ext-vector, so bit_cast from that exact type.
static __device__ __forceinline__ unsigned int pk16(float a, float b) {
  typedef __attribute__((ext_vector_type(2))) __fp16 fp16x2;
  fp16x2 h = __builtin_amdgcn_cvt_pkrtz(a, b);
  return __builtin_bit_cast(unsigned int, h);
}

// Fast transcendentals: v_exp_f32 + single v_rcp_f32.
static __device__ __forceinline__ float sigmoid_f(float x) {
  return __builtin_amdgcn_rcpf(1.0f + __expf(-x));
}
static __device__ __forceinline__ float tanh_f(float x) {
  return 1.0f - 2.0f * __builtin_amdgcn_rcpf(1.0f + __expf(2.0f * x));
}

#define NT    256
#define WAVES 8
#define LROW  20   // LDS transpose row stride (floats): 16B-aligned rows, conflict-free

__launch_bounds__(NT)
__global__ void gru_dense_softmax_wmma(
    const float* __restrict__ x,     // [B,11,3]
    const float* __restrict__ kern,  // [3,33]
    const float* __restrict__ reck,  // [11,33]
    const float* __restrict__ bias,  // [2,33] row0 = input bias, row1 = recurrent bias
    const float* __restrict__ dw,    // [121,7]
    const float* __restrict__ db,    // [7]
    float* __restrict__ out,         // [B,7] softmax probs
    long long Btot)
{
  constexpr int T = 11, F = 3, H = 11, C = 7, H3 = 33;

  __shared__ float ldsT[WAVES][16 * LROW];  // per-wave transpose tile
  __shared__ v8u   ldsD[T * 32];            // dense-head B tiles, B-register layout

  const int tid  = threadIdx.x;
  const int lane = tid & 31;
  const int wave = tid >> 5;
  const int n    = lane & 15;   // B-operand column / C-operand N index
  const int hi   = lane >> 4;   // A/B K-half select

  // ---- Dense-head B tiles (elem e <-> K = e<8 ? hi*8+e : 16+hi*8+(e-8), col n). Rows >=12 are zero, so the
  // stale x features sitting at K=12..14 of the A operand are multiplied by zero.
  for (int t = wave; t < T; t += WAVES) {
    UA b;
    #pragma unroll
    for (int e = 0; e < 16; ++e) {
      const int k = (e < 8) ? (hi * 8 + e) : (16 + hi * 8 + (e - 8));
      float val = 0.0f;
      if (n < C) {
        if (k < H)                 val = dw[(t * H + k) * C + n];
        else if (k == H && t == 0) val = db[n];     // dense bias, once (t=0)
      }
      b.h[e] = (_Float16)val;
    }
    ldsD[t * 32 + lane] = b.uv;
  }

  // ---- Fused gate B tiles over the combined operand [h(0..10) | 1(11) | x(12..14)]:
  //   g=0 (z):  rows0-10 = Rz, row11 = b_in[z]+b_rec[z], rows12-14 = Wz
  //   g=1 (r):  rows0-10 = Rr, row11 = b_in[r]+b_rec[r], rows12-14 = Wr
  //   g=2 (xh): rows0-10 = 0,  row11 = b_in[h],          rows12-14 = Wh
  //   g=3 (rh): rows0-10 = Rh, row11 = b_rec[h],         rows12-14 = 0
  v16h gateB[4];
  #pragma unroll
  for (int g = 0; g < 4; ++g) {
    UA b;
    #pragma unroll
    for (int e = 0; e < 16; ++e) {
      const int k = (e < 8) ? (hi * 8 + e) : (16 + hi * 8 + (e - 8));
      float val = 0.0f;
      if (n < H) {
        const int col = (g == 0 ? 0 : (g == 1 ? H : 2 * H)) + n;
        if (k < H) {
          if (g != 2) val = reck[k * H3 + col];
        } else if (k == H) {
          if (g <= 1)      val = bias[col] + bias[H3 + col];
          else if (g == 2) val = bias[col];        // input bias, h gate
          else             val = bias[H3 + col];   // recurrent bias, h gate
        } else if (k >= 12 && k < 12 + F) {
          if (g != 3) val = kern[(k - 12) * H3 + col];
        }
      }
      b.h[e] = (_Float16)val;
    }
    gateB[g] = b.v;
  }
  __syncthreads();

  const long long b0   = ((long long)blockIdx.x * WAVES + wave) * 16;
  const long long brow = (b0 + n < Btot) ? (b0 + n) : (Btot - 1);
  const float* xrow = x + brow * (long long)(T * F);

  const v8f zero8 = {0.f, 0.f, 0.f, 0.f, 0.f, 0.f, 0.f, 0.f};
  v8f hD = zero8;      // h in C/D layout: elem v -> batch row v+8*hi, hidden j = n
  v8f logits = zero8;  // dense-head accumulator

  // Combined A operand. K map: lanes<16: e0..7 -> K0..7; lanes>=16: e0..7 -> K8..15.
  //   lanes<16 : dwords u0..u3 = h[0..7]
  //   lanes>=16: u0 = h[8],h[9]; u1 = h[10],1.0(bias); u2 = x0,x1; u3 = x2,0
  UA hau;
  #pragma unroll
  for (int e = 0; e < 8; ++e) hau.u[e] = 0u;
  if (hi) hau.u[1] = pk16(0.0f, 1.0f);   // bias one at K=11 for the first step

  float* Tb = &ldsT[wave][0];

  #pragma unroll 1   // keep the loop rolled: weight tiles stay register-resident
  for (int t = 0; t < T; ++t) {
    if (hi) {   // x features live at K=12..14 -> lanes 16..31, dwords u2/u3
      hau.u[2] = pk16(xrow[t * F + 0], xrow[t * F + 1]);
      hau.u[3] = pk16(xrow[t * F + 2], 0.0f);
    }
    const v16h A = hau.v;

    v8f zs = wmma_f16(A, gateB[0], zero8);   // x@Wz + h@Rz + biases
    v8f rs = wmma_f16(A, gateB[1], zero8);   // x@Wr + h@Rr + biases
    v8f xh = wmma_f16(A, gateB[2], zero8);   // x@Wh + b_in[h]
    v8f rh = wmma_f16(A, gateB[3], zero8);   // h@Rh + b_rec[h]

    UC uz, ur, uxh, urh, uh;
    uz.v = zs; ur.v = rs; uxh.v = xh; urh.v = rh; uh.v = hD;
    #pragma unroll
    for (int v = 0; v < 8; ++v) {
      const float z  = sigmoid_f(uz.f[v]);
      const float r  = sigmoid_f(ur.f[v]);
      const float hh = tanh_f(fmaf(r, urh.f[v], uxh.f[v]));
      uh.f[v] = fmaf(z, uh.f[v] - hh, hh);   // z*h + (1-z)*hh
    }
    hD = uh.v;

    // Transpose h (C/D -> A layout) through the per-wave LDS tile.
    #pragma unroll
    for (int v = 0; v < 8; ++v) Tb[(v + 8 * hi) * LROW + n] = uh.f[v];
    asm volatile("s_wait_dscnt 0" ::: "memory");   // same-wave DS ops are in-order

    if (!hi) {
      const float* r0 = &Tb[n * LROW];              // 16B-aligned: 2x ds_load_b128
      hau.u[0] = pk16(r0[0], r0[1]);
      hau.u[1] = pk16(r0[2], r0[3]);
      hau.u[2] = pk16(r0[4], r0[5]);
      hau.u[3] = pk16(r0[6], r0[7]);
    } else {
      const float* r1 = &Tb[n * LROW + 8];
      hau.u[0] = pk16(r1[0], r1[1]);                // h[8], h[9]
      hau.u[1] = pk16(r1[2], 1.0f);                 // h[10], bias one (K=11)
      // u2/u3 hold stale x; dense rows 12..15 are zero.
    }

    // Dense head: logits += h_t @ W_t.
    UA dbt; dbt.uv = ldsD[t * 32 + lane];
    logits = wmma_f16(hau.v, dbt.v, logits);
  }

  // ---- Softmax: stage logits through the transpose tile, one batch row per lane 0..15.
  { UC ul; ul.v = logits;
    if (n < C) {
      #pragma unroll
      for (int v = 0; v < 8; ++v) Tb[(v + 8 * hi) * LROW + n] = ul.f[v];
    }
  }
  asm volatile("s_wait_dscnt 0" ::: "memory");

  if (!hi && (b0 + n) < Btot) {
    float l[C];
    #pragma unroll
    for (int c = 0; c < C; ++c) l[c] = Tb[n * LROW + c];
    float m = l[0];
    #pragma unroll
    for (int c = 1; c < C; ++c) m = fmaxf(m, l[c]);
    float e[C], s = 0.0f;
    #pragma unroll
    for (int c = 0; c < C; ++c) { e[c] = __expf(l[c] - m); s += e[c]; }
    const float inv = __builtin_amdgcn_rcpf(s);
    float* o = out + (b0 + n) * (long long)C;
    #pragma unroll
    for (int c = 0; c < C; ++c) o[c] = e[c] * inv;
  }
}

extern "C" void kernel_launch(void* const* d_in, const int* in_sizes, int n_in,
                              void* d_out, int out_size, void* d_ws, size_t ws_size,
                              hipStream_t stream) {
  (void)n_in; (void)out_size; (void)d_ws; (void)ws_size;
  const float* x    = (const float*)d_in[0];
  const float* kern = (const float*)d_in[1];
  const float* reck = (const float*)d_in[2];
  const float* bias = (const float*)d_in[3];
  const float* dw   = (const float*)d_in[4];
  const float* db   = (const float*)d_in[5];
  float* out = (float*)d_out;

  const long long Btot = (long long)in_sizes[0] / 33;   // [B,11,3]
  const long long batchesPerBlock = 16LL * WAVES;        // 16 per wave, 8 waves
  const int grid = (int)((Btot + batchesPerBlock - 1) / batchesPerBlock);

  gru_dense_softmax_wmma<<<grid, NT, 0, stream>>>(x, kern, reck, bias, dw, db, out, Btot);
}
